// TransformerDecoder_36155034698432
// MI455X (gfx1250) — compile-verified, matrix operand order checked by reading, not run on hardware
//
#include <hip/hip_runtime.h>
#include <hip/hip_bf16.h>

// ---------------------------------------------------------------------------
// GPT-style decoder forward for MI455X (gfx1250).
// WMMA f16-in / f32-acc GEMMs with a double-buffered Tensor-Data-Mover
// pipeline (tensor_load_to_lds + s_wait_tensorcnt), wave32 shuffle softmax,
// attention maps streamed once (the run is output-bandwidth bound: ~1 GB of
// attn maps + logits vs ~250 GFLOP of WMMA math).
// ---------------------------------------------------------------------------

#define BB 4
#define TT 1024
#define EE 512
#define HH 8
#define LL 4
#define VV 32000
#define HID 100
#define HIDP 128      // HID padded to a multiple of 32 for WMMA K-steps
#define HD 64         // head dim

#define BN 128        // GEMM block tile: 64 rows x 128 cols, 4 waves

typedef __attribute__((ext_vector_type(16))) _Float16 v16h;
typedef __attribute__((ext_vector_type(8)))  float    v8f;
typedef __attribute__((ext_vector_type(4)))  unsigned int u32x4;
typedef __attribute__((ext_vector_type(8)))  int      i32x8;
typedef __attribute__((ext_vector_type(4)))  int      i32x4;

// ROCm 7.2 (clang-22) exposes the 5-arg tensor_load_to_lds builtin and does
// NOT ship the TDM header; therock-10.0 ships the header and uses a 6-arg
// form. Use header presence to select the portable path.
#if defined(__has_builtin)
#if __has_builtin(__builtin_amdgcn_tensor_load_to_lds) && \
    !__has_include(<hip/amd_detail/amd_gfx1250_TDM.h>)
#define USE_TDM 1
#endif
#endif

// ---------------------------------------------------------------------------
// WMMA helpers (V_WMMA_F32_16X16X32_F16)
// ---------------------------------------------------------------------------
__device__ __forceinline__ v8f wmma_f16(v16h a, v16h b, v8f c) {
  return __builtin_amdgcn_wmma_f32_16x16x32_f16(
      /*neg_a=*/false, a, /*neg_b=*/false, b,
      /*c_mod=*/(short)0, c, /*reuse_a=*/false, /*reuse_b=*/false);
}

// 16-bit A fragment (16x32), ISA 7.12.2 layout:
//   lanes 0-15: row m=lane,    halfs 0-7 -> K=0..7,  halfs 8-15 -> K=16..23
//   lanes16-31: row m=lane-16, halfs 0-7 -> K=8..15, halfs 8-15 -> K=24..31
// The B operand of Q*K^T (K-matrix transposed) shares this per-lane pattern.
__device__ __forceinline__ v16h load_a_frag(const _Float16* A, int lda,
                                            int row0, int k0, int lane) {
  int m    = lane & 15;
  int base = (lane & 16) ? 8 : 0;
  const _Float16* p = A + (size_t)(row0 + m) * lda + k0 + base;
  v16h a;
#pragma unroll
  for (int j = 0; j < 8; ++j)  a[j] = p[j];
#pragma unroll
  for (int j = 8; j < 16; ++j) a[j] = p[j + 8];
  return a;
}

// Row-major B (K x N): lane holds column n = lane&15, same K mapping.
__device__ __forceinline__ v16h load_b_frag(const _Float16* B, int ldb,
                                            int k0, int col0, int lane) {
  int n    = lane & 15;
  int base = (lane & 16) ? 8 : 0;
  const _Float16* p = B + (size_t)(k0 + base) * ldb + col0 + n;
  v16h b;
#pragma unroll
  for (int j = 0; j < 8; ++j)  b[j] = p[(size_t)j * ldb];
#pragma unroll
  for (int j = 8; j < 16; ++j) b[j] = p[(size_t)(j + 8) * ldb];
  return b;
}

#if defined(USE_TDM)
// Issue one TDM transfer: 32 x BN f16 tile of row-major B(K x N) -> LDS.
// D# built per ISA ch.8: group0 = {count, lds_addr, global_addr, type},
// group1 = {data_size, tensor dims, tile dims, dim0 stride}.
__device__ __forceinline__ void tdm_load_btile(const _Float16* gsrc,
                                               unsigned lds_off, int N, int K) {
  unsigned long long ga = (unsigned long long)(uintptr_t)gsrc;
  u32x4 g0;
  g0[0] = 1u;                                   // count=1 (valid user D#)
  g0[1] = lds_off;                              // lds_addr (bytes)
  g0[2] = (unsigned)(ga & 0xFFFFFFFFu);         // global_addr[31:0]
  g0[3] = (unsigned)((ga >> 32) & 0x1FFFFFFu)   // global_addr[56:32]
          | (2u << 30);                         // type=2 ("image")
  i32x8 g1;
  g1[0] = 1 << 16;                              // data_size=1 (2 bytes)
  g1[1] = (int)((unsigned)(N & 0xFFFF) << 16);  // tensor_dim0[15:0]
  g1[2] = (int)(((unsigned)N >> 16) |
                ((unsigned)(K & 0xFFFF) << 16));// td0[31:16] | td1[15:0]
  g1[3] = (int)(((unsigned)K >> 16) |
                ((unsigned)BN << 16));          // td1[31:16] | tile_dim0=BN
  g1[4] = 32;                                   // tile_dim1=32, tile_dim2=0
  g1[5] = N;                                    // tensor_dim0_stride[31:0]
  g1[6] = 0;
  g1[7] = 0;
  i32x4 gz = {0, 0, 0, 0};
  __builtin_amdgcn_tensor_load_to_lds(g0, g1, gz, gz, 0);
}
#endif

// ---------------------------------------------------------------------------
// Elementwise kernels
// ---------------------------------------------------------------------------
__global__ void embed_kernel(const int* __restrict__ idx,
                             const float* __restrict__ tok,
                             const float* __restrict__ pos,
                             float* __restrict__ x) {
  size_t i = (size_t)blockIdx.x * blockDim.x + threadIdx.x;
  size_t n = (size_t)BB * TT * EE;
  if (i >= n) return;
  int e  = (int)(i % EE);
  int bt = (int)(i / EE);
  int t  = bt % TT;
  x[i] = tok[(size_t)idx[bt] * EE + e] + pos[(size_t)t * EE + e];
}

__global__ void cvt_f16_kernel(const float* __restrict__ src,
                               _Float16* __restrict__ dst, size_t n) {
  size_t i = (size_t)blockIdx.x * blockDim.x + threadIdx.x;
  for (; i < n; i += (size_t)gridDim.x * blockDim.x)
    dst[i] = (_Float16)src[i];
}

// Zero-padded 2D copy f32 -> f16: dst is Rp x Cp, src is R x C.
__global__ void pad2d_f16_kernel(const float* __restrict__ src, int R, int C,
                                 _Float16* __restrict__ dst, int Rp, int Cp) {
  size_t i = (size_t)blockIdx.x * blockDim.x + threadIdx.x;
  size_t n = (size_t)Rp * Cp;
  if (i >= n) return;
  int c = (int)(i % Cp), r = (int)(i / Cp);
  dst[i] = (r < R && c < C) ? (_Float16)src[(size_t)r * C + c] : (_Float16)0.0f;
}

__global__ void pad_bias_kernel(const float* __restrict__ src, int C,
                                float* __restrict__ dst, int Cp) {
  int i = blockIdx.x * blockDim.x + threadIdx.x;
  if (i < Cp) dst[i] = (i < C) ? src[i] : 0.0f;
}

// ---------------------------------------------------------------------------
// LayerNorm: one 256-thread block per row (E=512). Emits f16 activations.
// ---------------------------------------------------------------------------
__global__ __launch_bounds__(256) void ln_kernel(const float* __restrict__ x,
                                                 const float* __restrict__ g,
                                                 const float* __restrict__ b,
                                                 _Float16* __restrict__ out) {
  __shared__ float red[256];
  int row = blockIdx.x, tid = threadIdx.x;
  const float* xr = x + (size_t)row * EE;

  float s = 0.0f;
  for (int e = tid; e < EE; e += 256) s += xr[e];
  red[tid] = s; __syncthreads();
  for (int o = 128; o > 0; o >>= 1) { if (tid < o) red[tid] += red[tid + o]; __syncthreads(); }
  float mean = red[0] * (1.0f / EE); __syncthreads();

  float v = 0.0f;
  for (int e = tid; e < EE; e += 256) { float d = xr[e] - mean; v += d * d; }
  red[tid] = v; __syncthreads();
  for (int o = 128; o > 0; o >>= 1) { if (tid < o) red[tid] += red[tid + o]; __syncthreads(); }
  float rstd = rsqrtf(red[0] * (1.0f / EE) + 1e-5f);

  _Float16* orow = out + (size_t)row * EE;
  for (int e = tid; e < EE; e += 256)
    orow[e] = (_Float16)((xr[e] - mean) * rstd * g[e] + b[e]);
}

// ---------------------------------------------------------------------------
// WMMA GEMM: out = A(MxK,f16) @ B(KxN,f16) [+bias][+resid][relu]
//   Block = 128 threads (4 waves) -> 64 x 128 output tile.
//   Wave w owns rows [row0+16w, +16), all 128 cols: 8 accumulators, one
//   A-fragment feeds 8 WMMAs per K-step.
//   B K-step tiles (32 x 128 f16 = 8KB) are staged in LDS through a
//   DOUBLE-BUFFERED Tensor Data Mover pipeline: wave 0 issues the DMA for
//   step s+1 before computing step s, then waits s_wait_tensorcnt<=1
//   (TDM completions are in-order per wave => current tile has landed).
//   Falls back to cooperative ds staging on the 6-arg-TDM toolchain.
//   Requires M%64==0, N%128==0, K%32==0 (true for all uses here).
// ---------------------------------------------------------------------------
template <bool HAS_BIAS, bool HAS_RES, bool DO_RELU, bool OUT_F, bool OUT_H>
__global__ __launch_bounds__(128) void gemm_wmma_kernel(
    const _Float16* __restrict__ A, const _Float16* __restrict__ Bm,
    int M, int N, int K,
    const float* __restrict__ bias, const float* __restrict__ resid,
    float* __restrict__ outF, _Float16* __restrict__ outH) {
  __shared__ _Float16 btile[2][32 * BN];   // double-buffered B stage (16KB)

  const int wave = threadIdx.x >> 5;
  const int lane = threadIdx.x & 31;
  const int col0 = blockIdx.x * BN;
  const int row0 = blockIdx.y * 64 + wave * 16;
  const int nsteps = K >> 5;

  v8f acc[8] = {{}, {}, {}, {}, {}, {}, {}, {}};

  for (int s = 0; s < nsteps; ++s) {
#if defined(USE_TDM)
    if (threadIdx.x < 32) {   // wave 0 drives the DMA pipeline
      if (s == 0)
        tdm_load_btile(Bm + (size_t)0 * N + col0,
                       (unsigned)(uintptr_t)(void*)btile[0], N, K);
      if (s + 1 < nsteps) {
        tdm_load_btile(Bm + (size_t)(s + 1) * 32 * N + col0,
                       (unsigned)(uintptr_t)(void*)btile[(s + 1) & 1], N, K);
        __builtin_amdgcn_s_wait_tensorcnt(1);   // tile s has landed
      } else {
        __builtin_amdgcn_s_wait_tensorcnt(0);
      }
    }
    __syncthreads();
#else
    // cooperative fallback: 128 threads x 32 halfs = 32 x 128 tile
    {
      const _Float16* src = Bm + (size_t)s * 32 * N + col0;
      int r = threadIdx.x >> 2;          // 0..31 (K rows)
      int c = (threadIdx.x & 3) * 32;    // 0,32,64,96
#pragma unroll
      for (int j = 0; j < 32; ++j)
        btile[s & 1][r * BN + c + j] = src[(size_t)r * N + c + j];
    }
    __syncthreads();
#endif

    // prefetch this wave's next A segment (speculative; dropped if OOB)
    __builtin_prefetch(A + (size_t)(row0 + (lane & 15)) * K + s * 32 + 32, 0, 1);

    const _Float16* bt = btile[s & 1];
    v16h a = load_a_frag(A, K, row0, s * 32, lane);
#pragma unroll
    for (int nt = 0; nt < 8; ++nt) {
      v16h b = load_b_frag(bt, BN, 0, nt * 16, lane);
      acc[nt] = wmma_f16(a, b, acc[nt]);
    }
    __syncthreads();   // all waves done with btile[s&1] before step s+2 DMA
  }

  const int n  = lane & 15;
  const int rb = ((lane >> 4) & 1) * 8;
#pragma unroll
  for (int nt = 0; nt < 8; ++nt) {
    const int col = col0 + nt * 16 + n;
    float bv = HAS_BIAS ? bias[col] : 0.0f;
#pragma unroll
    for (int i = 0; i < 8; ++i) {
      const int r = row0 + rb + i;
      float v = acc[nt][i] + bv;
      if (HAS_RES) v += resid[(size_t)r * N + col];
      if (DO_RELU) v = fmaxf(v, 0.0f);
      if (OUT_F)   outF[(size_t)r * N + col] = v;
      if (OUT_H)   outH[(size_t)r * N + col] = (_Float16)v;
    }
  }
}

// ---------------------------------------------------------------------------
// Attention: one wave handles 16 query rows for one (b,h).
//  scores (QK^T) via WMMA -> LDS, causal softmax (wave32 shuffles),
//  attention map streamed to d_out, then wei@V via WMMA -> y16.
// ---------------------------------------------------------------------------
__global__ __launch_bounds__(32) void attn_kernel(
    const _Float16* __restrict__ q16, const _Float16* __restrict__ k16,
    const _Float16* __restrict__ v16, _Float16* __restrict__ y16,
    float* __restrict__ attn_out, int layer) {
  __shared__ float    sc[16 * TT];   // 64 KB score/prob staging
  __shared__ _Float16 wf[16 * TT];   // 32 KB f16 probs (96KB < 320KB WGP LDS)

  int bh   = blockIdx.x;             // b*H + h
  int b    = bh / HH;
  int h    = bh % HH;
  int qb   = blockIdx.y;             // query 16-row block
  int lane = threadIdx.x;
  int q0   = qb * 16;
  const float scale = 0.125f;        // 1/sqrt(64)

  const _Float16* qp = q16 + (size_t)b * TT * EE + h * HD;
  const _Float16* kp = k16 + (size_t)b * TT * EE + h * HD;
  const _Float16* vp = v16 + (size_t)b * TT * EE + h * HD;

  // Q tile fragments: 16 x 64 => two K=32 fragments
  v16h a0 = load_a_frag(qp, EE, q0, 0, lane);
  v16h a1 = load_a_frag(qp, EE, q0, 32, lane);

  int nkb   = qb + 1;
  int nkeys = nkb * 16;

  // ---- scores = Q @ K^T  (K^T fragment == A-style layout over key rows) ----
  for (int kb = 0; kb < nkb; ++kb) {
    int key0 = kb * 16;
    v16h b0 = load_a_frag(kp, EE, key0, 0, lane);
    v16h b1 = load_a_frag(kp, EE, key0, 32, lane);
    v8f acc = {};
    acc = wmma_f16(a0, b0, acc);
    acc = wmma_f16(a1, b1, acc);
    int n  = lane & 15;
    int rb = ((lane >> 4) & 1) * 8;
#pragma unroll
    for (int i = 0; i < 8; ++i)
      sc[(rb + i) * TT + key0 + n] = acc[i] * scale;
  }
  __syncthreads();

  // ---- causal softmax per row; stream attention map; stash f16 probs ----
  float* arow_base =
      attn_out + (((size_t)layer * BB + b) * HH + h) * (size_t)TT * TT + (size_t)q0 * TT;
  for (int r = 0; r < 16; ++r) {
    int grow = q0 + r;
    float mx = -3.0e38f;
    for (int j = lane; j < nkeys; j += 32)
      if (j <= grow) mx = fmaxf(mx, sc[r * TT + j]);
#pragma unroll
    for (int o = 16; o; o >>= 1) mx = fmaxf(mx, __shfl_xor(mx, o, 32));

    float sum = 0.0f;
    for (int j = lane; j < nkeys; j += 32) {
      float p = (j <= grow) ? __expf(sc[r * TT + j] - mx) : 0.0f;
      sc[r * TT + j] = p;
      sum += p;
    }
#pragma unroll
    for (int o = 16; o; o >>= 1) sum += __shfl_xor(sum, o, 32);
    float inv = 1.0f / sum;

    float* arow = arow_base + (size_t)r * TT;
    for (int j = lane; j < TT; j += 32) {
      float w = (j < nkeys) ? sc[r * TT + j] * inv : 0.0f;
      arow[j] = w;                               // masked entries written as 0
      if (j < nkeys) wf[r * TT + j] = (_Float16)w;
    }
  }
  __syncthreads();

  // ---- y = wei @ V  (HD=64 -> 4 column tiles) ----
  v8f yacc[4] = {{}, {}, {}, {}};
  for (int k0 = 0; k0 < nkeys; k0 += 32) {
    v16h aw = load_a_frag(wf, TT, 0, k0, lane);
#pragma unroll
    for (int nt = 0; nt < 4; ++nt) {
      v16h bv = load_b_frag(vp, EE, k0, nt * 16, lane);
      yacc[nt] = wmma_f16(aw, bv, yacc[nt]);
    }
  }
  int n  = lane & 15;
  int rb = ((lane >> 4) & 1) * 8;
#pragma unroll
  for (int nt = 0; nt < 4; ++nt)
#pragma unroll
    for (int i = 0; i < 8; ++i)
      y16[((size_t)b * TT + q0 + rb + i) * EE + h * HD + nt * 16 + n] =
          (_Float16)yacc[nt][i];
}

// ---------------------------------------------------------------------------
// Host-side orchestration
// ---------------------------------------------------------------------------
static inline void* ws_carve(char*& p, size_t bytes) {
  void* r = (void*)p;
  p += (bytes + 255) & ~(size_t)255;
  return r;
}

extern "C" void kernel_launch(void* const* d_in, const int* in_sizes, int n_in,
                              void* d_out, int out_size, void* d_ws, size_t ws_size,
                              hipStream_t stream) {
  const int*   idx    = (const int*)  d_in[0];
  const float* tok    = (const float*)d_in[1];
  const float* pos    = (const float*)d_in[2];
  const float* ln1_g  = (const float*)d_in[3];
  const float* ln1_b  = (const float*)d_in[4];
  const float* Wq     = (const float*)d_in[5];
  const float* Wk     = (const float*)d_in[6];
  const float* Wv     = (const float*)d_in[7];
  const float* Wo     = (const float*)d_in[8];
  const float* bo     = (const float*)d_in[9];
  const float* ln2_g  = (const float*)d_in[10];
  const float* ln2_b  = (const float*)d_in[11];
  const float* W1     = (const float*)d_in[12];
  const float* b1     = (const float*)d_in[13];
  const float* W2     = (const float*)d_in[14];
  const float* b2     = (const float*)d_in[15];
  const float* lnf_g  = (const float*)d_in[16];
  const float* lnf_b  = (const float*)d_in[17];
  const float* Wlm    = (const float*)d_in[18];

  float* out_logits = (float*)d_out;                              // (B,T,V)
  float* out_attn   = out_logits + (size_t)BB * TT * VV;          // (L,B,H,T,T)

  // ---- workspace layout ----
  char* p = (char*)d_ws;
  const size_t NTOK = (size_t)BB * TT;
  float*    x     = (float*)   ws_carve(p, NTOK * EE * 4);
  _Float16* h16   = (_Float16*)ws_carve(p, NTOK * EE * 2);
  _Float16* q16   = (_Float16*)ws_carve(p, NTOK * EE * 2);
  _Float16* k16   = (_Float16*)ws_carve(p, NTOK * EE * 2);
  _Float16* v16   = (_Float16*)ws_carve(p, NTOK * EE * 2);
  _Float16* y16   = (_Float16*)ws_carve(p, NTOK * EE * 2);
  _Float16* hid16 = (_Float16*)ws_carve(p, NTOK * HIDP * 2);
  _Float16* wq16  = (_Float16*)ws_carve(p, (size_t)EE * EE * 2);
  _Float16* wk16  = (_Float16*)ws_carve(p, (size_t)EE * EE * 2);
  _Float16* wv16  = (_Float16*)ws_carve(p, (size_t)EE * EE * 2);
  _Float16* wo16  = (_Float16*)ws_carve(p, (size_t)EE * EE * 2);
  _Float16* w1p16 = (_Float16*)ws_carve(p, (size_t)EE * HIDP * 2);
  _Float16* w2p16 = (_Float16*)ws_carve(p, (size_t)HIDP * EE * 2);
  float*    b1p   = (float*)   ws_carve(p, HIDP * 4);
  _Float16* wlm16 = (_Float16*)ws_carve(p, (size_t)EE * VV * 2);
  (void)ws_size; (void)n_in; (void)in_sizes; (void)out_size;

  // ---- one-time conversions ----
  cvt_f16_kernel<<<4096, 256, 0, stream>>>(Wlm, wlm16, (size_t)EE * VV);

  // ---- embedding ----
  {
    size_t n = NTOK * EE;
    embed_kernel<<<(unsigned)((n + 255) / 256), 256, 0, stream>>>(idx, tok, pos, x);
  }

  const size_t EV = (size_t)EE * EE;
  const dim3 gemmE(EE / BN,   (unsigned)(NTOK / 64));    // N=512 GEMMs
  const dim3 gemmH(HIDP / BN, (unsigned)(NTOK / 64));    // N=128 GEMM (FFN1)
  const dim3 gemmV(VV / BN,   (unsigned)(NTOK / 64));    // N=32000 GEMM (logits)

  for (int l = 0; l < LL; ++l) {
    // ln1 -> h16
    ln_kernel<<<(unsigned)NTOK, 256, 0, stream>>>(x, ln1_g + (size_t)l * EE,
                                                  ln1_b + (size_t)l * EE, h16);
    // weight conversions for this layer
    cvt_f16_kernel<<<1024, 256, 0, stream>>>(Wq + l * EV, wq16, EV);
    cvt_f16_kernel<<<1024, 256, 0, stream>>>(Wk + l * EV, wk16, EV);
    cvt_f16_kernel<<<1024, 256, 0, stream>>>(Wv + l * EV, wv16, EV);
    cvt_f16_kernel<<<1024, 256, 0, stream>>>(Wo + l * EV, wo16, EV);

    // q,k,v = h @ W (f16 out, no bias)
    gemm_wmma_kernel<false,false,false,false,true><<<gemmE, 128, 0, stream>>>(
        h16, wq16, (int)NTOK, EE, EE, nullptr, nullptr, nullptr, q16);
    gemm_wmma_kernel<false,false,false,false,true><<<gemmE, 128, 0, stream>>>(
        h16, wk16, (int)NTOK, EE, EE, nullptr, nullptr, nullptr, k16);
    gemm_wmma_kernel<false,false,false,false,true><<<gemmE, 128, 0, stream>>>(
        h16, wv16, (int)NTOK, EE, EE, nullptr, nullptr, nullptr, v16);

    // attention: scores, softmax, attn-map dump, wei@V
    attn_kernel<<<dim3(BB * HH, TT / 16), 32, 0, stream>>>(q16, k16, v16, y16,
                                                           out_attn, l);

    // x = x + y @ Wo + bo   (f32 out, in-place residual)
    gemm_wmma_kernel<true,true,false,true,false><<<gemmE, 128, 0, stream>>>(
        y16, wo16, (int)NTOK, EE, EE, bo + (size_t)l * EE, x, x, nullptr);

    // ln2 -> h16
    ln_kernel<<<(unsigned)NTOK, 256, 0, stream>>>(x, ln2_g + (size_t)l * EE,
                                                  ln2_b + (size_t)l * EE, h16);

    // FFN1: hid = relu(h @ W1p + b1p), padded N=128
    pad2d_f16_kernel<<<(EE * HIDP + 255) / 256, 256, 0, stream>>>(
        W1 + (size_t)l * EE * HID, EE, HID, w1p16, EE, HIDP);
    pad_bias_kernel<<<1, HIDP, 0, stream>>>(b1 + (size_t)l * HID, HID, b1p, HIDP);
    gemm_wmma_kernel<true,false,true,false,true><<<gemmH, 128, 0, stream>>>(
        h16, w1p16, (int)NTOK, HIDP, EE, b1p, nullptr, nullptr, hid16);

    // FFN2: x = x + hid @ W2p + b2, padded K=128 (pad rows are zero)
    pad2d_f16_kernel<<<(HIDP * EE + 255) / 256, 256, 0, stream>>>(
        W2 + (size_t)l * HID * EE, HID, EE, w2p16, HIDP, EE);
    gemm_wmma_kernel<true,true,false,true,false><<<gemmE, 128, 0, stream>>>(
        hid16, w2p16, (int)NTOK, EE, HIDP, b2 + (size_t)l * EE, x, x, nullptr);
  }

  // final LN + logits
  ln_kernel<<<(unsigned)NTOK, 256, 0, stream>>>(x, lnf_g, lnf_b, h16);
  gemm_wmma_kernel<false,false,false,true,false><<<gemmV, 128, 0, stream>>>(
      h16, wlm16, (int)NTOK, VV, EE, nullptr, nullptr, out_logits, nullptr);
}
